// Model_75204877353794
// MI455X (gfx1250) — matrix operand order repaired and checked by reading, not compiled, not run on hardware
//
#include <hip/hip_runtime.h>
#include <hip/hip_bf16.h>

typedef __attribute__((ext_vector_type(16))) _Float16 v16h;
typedef __attribute__((ext_vector_type(4)))  _Float16 v4h;
typedef __attribute__((ext_vector_type(8)))  float    v8f;

#define PCOL 20          // padded patch row stride (18 needed)
#define KDIM 2304        // 256*9 reduction dim
#define NKC  72          // KDIM/32 chunks
#define PATCH_H (256 * 3 * PCOL)        // 15360 halves
#define ATILE_H (NKC * 32 * 16)         // 36864 halves
#define SMEM_BYTES ((PATCH_H + ATILE_H) * 2)   // 104448 B dynamic LDS
#define WPB 8            // waves per block
#define CAP  2048        // candidate capacity per level per image
#define NBIN 4096
#define BATCH 4

// ---------------------------------------------------------------- utilities
__global__ void k_zero32(unsigned* __restrict__ p, size_t n) {
  size_t i = (size_t)blockIdx.x * blockDim.x + threadIdx.x;
  if (i < n) p[i] = 0u;
}

__global__ void k_f32_to_f16(const float* __restrict__ in, _Float16* __restrict__ out, size_t n) {
  size_t i = (size_t)blockIdx.x * blockDim.x + threadIdx.x;
  if (i < n) out[i] = (_Float16)in[i];
}

// convert [rows_real][rowlen] -> [rows_pad][rowlen], zero-filling padded rows
__global__ void k_f32_to_f16_padrows(const float* __restrict__ in, _Float16* __restrict__ out,
                                     int rows_real, int rows_pad, int rowlen) {
  size_t i = (size_t)blockIdx.x * blockDim.x + threadIdx.x;
  size_t n = (size_t)rows_pad * rowlen;
  if (i >= n) return;
  int row = (int)(i / rowlen);
  out[i] = (row < rows_real) ? (_Float16)in[i] : (_Float16)0.0f;
}

// ------------------------------------------------- conv3x3 via WMMA (f16 in, f16 out)
// X:[B,256,H,W] f16   Wt:[CoutPad,2304] f16 (k = c*9 + ky*3 + kx)   bias:[Cout] f32
// Block = 8 waves; block covers 16 pixels (one x-tile of one row) x 128 out-channels.
// Stage 1: halo patch -> LDS.  Stage 2: im2col pre-swizzled into WMMA A-fragment
// order aTile[kc][lane][h] so the main loop is pure ds_load_b128 + global_load_b128
// + v_wmma_f32_16x16x32_f16.
__global__ __launch_bounds__(32 * WPB)
void conv3x3_wmma(const _Float16* __restrict__ X,
                  const _Float16* __restrict__ Wt,
                  const float* __restrict__ bias,
                  _Float16* __restrict__ Y,
                  int Cout, int CoutPad, int H, int W, int tilesX, int relu)
{
  extern __shared__ _Float16 smem[];
  _Float16* patch = smem;                 // [256][3][PCOL]
  _Float16* aTile = smem + PATCH_H;       // [NKC][32][16]

  int tile = blockIdx.x;
  int tx = tile % tilesX;
  int t2 = tile / tilesX;
  int y  = t2 % H;
  int b  = t2 / H;
  int x0 = tx * 16;

  // ---- stage 1: cooperative halo patch load (rows y-1..y+1, cols x0-1..x0+16)
  for (int i = threadIdx.x; i < PATCH_H; i += 32 * WPB) {
    int col = i % PCOL;
    int rr  = (i / PCOL) % 3;
    int ch  = i / (3 * PCOL);
    int xx = x0 - 1 + col;
    int yy = y - 1 + rr;
    _Float16 v = (_Float16)0.0f;
    if (col < 18 && xx >= 0 && xx < W && yy >= 0 && yy < H)
      v = X[((size_t)(b * 256 + ch) * H + yy) * W + xx];
    patch[i] = v;
  }
  __syncthreads();

  // ---- stage 2: im2col swizzle into A-fragment order (contiguous k, same as B).
  //   aTile[kc*512 + lane*16 + h] = patch value for pixel m=lane&15,
  //   K(lane,h) = (lane>>4)*16 + h, k = kc*32 + K.
  for (int t0 = threadIdx.x * 16; t0 < ATILE_H; t0 += 32 * WPB * 16) {
    int lane16 = t0 >> 4;
    int lane = lane16 & 31;
    int kc   = lane16 >> 5;
    int m  = lane & 15;
    int hi = lane >> 4;
    v16h v;
#pragma unroll
    for (int h = 0; h < 16; ++h) {
      int K  = (hi << 4) + h;
      int k  = kc * 32 + K;
      int c  = k / 9;
      int j  = k - c * 9;
      int ky = j / 3;
      int kx = j - ky * 3;
      v[h] = patch[c * (3 * PCOL) + ky * PCOL + m + kx];
    }
    *(v16h*)(aTile + t0) = v;
  }
  __syncthreads();

  // ---- main loop: one wave = 16 pixels x 16 out-channels
  int lane = threadIdx.x & 31;
  int wv   = threadIdx.x >> 5;
  int ntile  = blockIdx.y * WPB + wv;
  int ntiles = CoutPad >> 4;
  if (ntile >= ntiles) return;          // wave-uniform exit (EXEC all-ones for WMMA)

  int hi = lane >> 4;
  int n  = ntile * 16 + (lane & 15);    // out channel held by this lane (B column)

  const _Float16* wrow = Wt + (size_t)n * KDIM + hi * 16;  // B frag: K = hi*16 + h
  const _Float16* arow = aTile + lane * 16;
  v8f acc = {};

#pragma unroll 4
  for (int kc = 0; kc < NKC; ++kc) {
    __builtin_prefetch(wrow + kc * 32 + 256, 0, 0);        // global_prefetch_b8
    v16h a = *(const v16h*)(arow + kc * 512);
    v16h bf = *(const v16h*)(wrow + kc * 32);
    acc = __builtin_amdgcn_wmma_f32_16x16x32_f16(false, a, false, bf,
                                                 (short)0, acc, false, false);
  }

  if (n < Cout) {
    float bv = bias[n];
    _Float16* dst = Y + ((size_t)(b * Cout + n) * H + y) * W + x0 + hi * 8;
    if (x0 + 16 <= W) {
      // full tile: branch-free packed stores (8B aligned for W in {80,40,20})
      v4h p0, p1;
#pragma unroll
      for (int v = 0; v < 4; ++v) {
        float o = acc[v] + bv;
        if (relu) o = fmaxf(o, 0.0f);
        p0[v] = (_Float16)o;
      }
#pragma unroll
      for (int v = 4; v < 8; ++v) {
        float o = acc[v] + bv;
        if (relu) o = fmaxf(o, 0.0f);
        p1[v - 4] = (_Float16)o;
      }
      *(v4h*)dst = p0;
      *(v4h*)(dst + 4) = p1;
    } else {
#pragma unroll
      for (int v = 0; v < 8; ++v) {
        int xx = x0 + hi * 8 + v;
        if (xx < W) {
          float o = acc[v] + bv;
          if (relu) o = fmaxf(o, 0.0f);
          dst[v] = (_Float16)o;
        }
      }
    }
  }
}

// ---------------------------------------------------------------- decode / top-k
__global__ void k_hist(const _Float16* __restrict__ cls_raw, int N1, unsigned* __restrict__ hist) {
  size_t i = (size_t)blockIdx.x * blockDim.x + threadIdx.x;
  if (i >= (size_t)BATCH * N1) return;
  int img = (int)(i / N1);
  float l = (float)cls_raw[i];
  float s = 1.0f / (1.0f + expf(-l));
  if (s >= 0.05f) {
    int bin = (int)(s * (float)NBIN);
    if (bin > NBIN - 1) bin = NBIN - 1;
    atomicAdd(&hist[img * NBIN + bin], 1u);
  }
}

__global__ void k_cutoff(const unsigned* __restrict__ hist, int* __restrict__ cutoff) {
  int img = threadIdx.x;
  if (img >= BATCH) return;
  const unsigned* h = hist + img * NBIN;
  int cnt = 0, cb = 0;
  for (int b = NBIN - 1; b >= 0; --b) {
    cnt += (int)h[b];
    if (cnt >= 1000) { cb = b; break; }
  }
  cutoff[img] = (cnt >= 1000) ? cb : 0;
}

__device__ inline void anchor4(int a, float stride, float* out) {
  const float ratios[3] = {1.0f, 2.0f, 0.5f};
  int i = a / 3, j = a - 3 * i;
  float scale = 4.0f * exp2f((float)i / 3.0f);
  float r  = ratios[j];
  float ws = sqrtf(stride * stride / r);
  float dw = ws, dh = ws * r;
  out[0] = 0.5f * (stride - dw * scale);
  out[1] = 0.5f * (stride - dh * scale);
  out[2] = 0.5f * (stride + dw * scale) - 1.0f;
  out[3] = 0.5f * (stride + dh * scale) - 1.0f;
}

__global__ void k_collect(const _Float16* __restrict__ cls_raw, const _Float16* __restrict__ box_raw,
                          const int* __restrict__ cutoff, int* __restrict__ cnt,
                          float* __restrict__ cand_s, float* __restrict__ cand_b,
                          float* __restrict__ cand_c,
                          int H, int W, int lvl, float stride) {
  int N1 = 720 * H * W;
  size_t i = (size_t)blockIdx.x * blockDim.x + threadIdx.x;
  if (i >= (size_t)BATCH * N1) return;
  int img = (int)(i / N1);
  int e   = (int)(i - (size_t)img * N1);
  float l = (float)cls_raw[i];
  float s = 1.0f / (1.0f + expf(-l));
  if (s < 0.05f) return;
  int bin = (int)(s * (float)NBIN);
  if (bin > NBIN - 1) bin = NBIN - 1;
  if (bin < cutoff[img]) return;
  int slot = atomicAdd(&cnt[img], 1);
  if (slot >= CAP) return;

  int x  = e % W;
  int t  = e / W;
  int yy = t % H;
  int ch = t / H;
  int c  = ch % 80;
  int a  = ch / 80;

  float d[4];
#pragma unroll
  for (int dd = 0; dd < 4; ++dd)
    d[dd] = (float)box_raw[((size_t)(img * 36 + a * 4 + dd) * H + yy) * W + x];

  float an[4]; anchor4(a, stride, an);
  float gx = x * stride, gy = yy * stride;
  float ax1 = gx + an[0], ay1 = gy + an[1], ax2 = gx + an[2], ay2 = gy + an[3];
  float w_ = ax2 - ax1 + 1.0f, h_ = ay2 - ay1 + 1.0f;
  float cx = ax1 + 0.5f * w_,  cy = ay1 + 0.5f * h_;
  float pcx = d[0] * w_ + cx,  pcy = d[1] * h_ + cy;
  float pw  = expf(d[2]) * w_, ph  = expf(d[3]) * h_;
  float Mx = (float)W * stride - 1.0f, My = (float)H * stride - 1.0f;
  float bx1 = fminf(fmaxf(pcx - 0.5f * (pw - 1.0f), 0.0f), Mx);
  float by1 = fminf(fmaxf(pcy - 0.5f * (ph - 1.0f), 0.0f), My);
  float bx2 = fminf(fmaxf(pcx + 0.5f * (pw - 1.0f), 0.0f), Mx);
  float by2 = fminf(fmaxf(pcy + 0.5f * (ph - 1.0f), 0.0f), My);

  int base = (img * 5 + lvl) * CAP + slot;
  cand_s[base] = s;
  cand_c[base] = (float)c;
  cand_b[base * 4 + 0] = bx1;
  cand_b[base * 4 + 1] = by1;
  cand_b[base * 4 + 2] = bx2;
  cand_b[base * 4 + 3] = by2;
}

// ---------------------------------------------------------------- NMS (one block/image)
__global__ __launch_bounds__(256)
void k_nms(float* __restrict__ sc_all, const float* __restrict__ cb_all,
           const float* __restrict__ cc_all, float* __restrict__ out) {
  int img = blockIdx.x;
  const int NC = 5 * CAP;
  float* sc = sc_all + (size_t)img * NC;
  const float* bx = cb_all + (size_t)img * NC * 4;
  const float* cl = cc_all + (size_t)img * NC;

  __shared__ float sv[256];
  __shared__ int   si[256];
  __shared__ float bestS;
  __shared__ int   bestJ;
  int tid = threadIdx.x;

  for (int det = 0; det < 100; ++det) {
    float best = 0.0f; int bj = -1;
    for (int i = tid; i < NC; i += 256) {
      float v = sc[i];
      if (v > best) { best = v; bj = i; }
    }
    sv[tid] = best; si[tid] = bj;
    __syncthreads();
    for (int s2 = 128; s2 > 0; s2 >>= 1) {
      if (tid < s2 && sv[tid + s2] > sv[tid]) { sv[tid] = sv[tid + s2]; si[tid] = si[tid + s2]; }
      __syncthreads();
    }
    if (tid == 0) { bestS = sv[0]; bestJ = si[0]; }
    __syncthreads();
    float s = bestS; int j = bestJ;

    if (s > 0.0f && j >= 0) {
      float b0 = bx[j * 4], b1 = bx[j * 4 + 1], b2 = bx[j * 4 + 2], b3 = bx[j * 4 + 3];
      float bc = cl[j];
      if (tid == 0) {
        int o = img * 100 + det;
        out[o] = s;
        out[400 + o * 4 + 0] = b0; out[400 + o * 4 + 1] = b1;
        out[400 + o * 4 + 2] = b2; out[400 + o * 4 + 3] = b3;
        out[2000 + o] = bc;
      }
      float areaj = (b2 - b0 + 1.0f) * (b3 - b1 + 1.0f);
      for (int i = tid; i < NC; i += 256) {
        float v = sc[i];
        if (v <= 0.0f) continue;
        if (i == j) { sc[i] = 0.0f; continue; }
        float x1 = fmaxf(bx[i * 4],     b0), y1 = fmaxf(bx[i * 4 + 1], b1);
        float x2 = fminf(bx[i * 4 + 2], b2), y2 = fminf(bx[i * 4 + 3], b3);
        float iw = fmaxf(x2 - x1 + 1.0f, 0.0f), ih = fmaxf(y2 - y1 + 1.0f, 0.0f);
        float inter = iw * ih;
        float ai = (bx[i * 4 + 2] - bx[i * 4] + 1.0f) * (bx[i * 4 + 3] - bx[i * 4 + 1] + 1.0f);
        float iou = inter / (ai + areaj - inter);
        if (iou > 0.5f && cl[i] == bc) sc[i] = 0.0f;
      }
    }
    __syncthreads();
  }
}

// ---------------------------------------------------------------- host driver
static inline size_t gridFor(size_t n, int bs) { return (n + (size_t)bs - 1) / bs; }

extern "C" void kernel_launch(void* const* d_in, const int* in_sizes, int n_in,
                              void* d_out, int out_size, void* d_ws, size_t ws_size,
                              hipStream_t stream) {
  (void)in_sizes; (void)n_in; (void)out_size; (void)ws_size;
  const float* feats[5] = {(const float*)d_in[0], (const float*)d_in[1], (const float*)d_in[2],
                           (const float*)d_in[3], (const float*)d_in[4]};
  const float* cls_wh = (const float*)d_in[5];
  const float* cls_bh = (const float*)d_in[6];
  const float* cls_wo = (const float*)d_in[7];
  const float* cls_bo = (const float*)d_in[8];
  const float* box_wh = (const float*)d_in[9];
  const float* box_bh = (const float*)d_in[10];
  const float* box_wo = (const float*)d_in[11];
  const float* box_bo = (const float*)d_in[12];

  char* ws = (char*)d_ws;
  size_t off = 0;
  auto alloc = [&](size_t bytes) { size_t o = off; off = (off + bytes + 255) & ~(size_t)255; return o; };

  const size_t WLAYER = 256UL * KDIM;                 // halves per hidden layer
  _Float16* w_cls_h = (_Float16*)(ws + alloc(4 * WLAYER * 2));
  _Float16* w_cls_o = (_Float16*)(ws + alloc(720UL * KDIM * 2));
  _Float16* w_box_h = (_Float16*)(ws + alloc(4 * WLAYER * 2));
  _Float16* w_box_o = (_Float16*)(ws + alloc(48UL * KDIM * 2));
  const size_t ACT = 4UL * 256 * 80 * 80;             // max level activation elems
  _Float16* act0 = (_Float16*)(ws + alloc(ACT * 2));
  _Float16* act1 = (_Float16*)(ws + alloc(ACT * 2));
  _Float16* act2 = (_Float16*)(ws + alloc(ACT * 2));
  _Float16* cls_raw = (_Float16*)(ws + alloc(4UL * 720 * 80 * 80 * 2));
  _Float16* box_raw = (_Float16*)(ws + alloc(4UL * 36 * 80 * 80 * 2));
  float* cand_s = (float*)(ws + alloc(4UL * 5 * CAP * 4));
  float* cand_b = (float*)(ws + alloc(4UL * 5 * CAP * 4 * 4));
  float* cand_c = (float*)(ws + alloc(4UL * 5 * CAP * 4));
  unsigned* hist = (unsigned*)(ws + alloc(4UL * NBIN * 4));   // hist, then cnt contiguous
  int* cnt    = (int*)(ws + alloc(4 * 4));
  int* cutoff = (int*)(ws + alloc(4 * 4));

  // allow >64KB dynamic LDS for the conv kernel (320KB/WGP on CDNA5)
  (void)hipFuncSetAttribute((const void*)conv3x3_wmma,
                            hipFuncAttributeMaxDynamicSharedMemorySize, SMEM_BYTES);

  // weight conversion (once per launch)
  k_f32_to_f16<<<gridFor(4 * WLAYER, 256), 256, 0, stream>>>(cls_wh, w_cls_h, 4 * WLAYER);
  k_f32_to_f16<<<gridFor(720UL * KDIM, 256), 256, 0, stream>>>(cls_wo, w_cls_o, 720UL * KDIM);
  k_f32_to_f16<<<gridFor(4 * WLAYER, 256), 256, 0, stream>>>(box_wh, w_box_h, 4 * WLAYER);
  k_f32_to_f16_padrows<<<gridFor(48UL * KDIM, 256), 256, 0, stream>>>(box_wo, w_box_o, 36, 48, KDIM);

  // zero outputs + candidate scores
  k_zero32<<<gridFor(2400, 256), 256, 0, stream>>>((unsigned*)d_out, 2400);
  k_zero32<<<gridFor(4UL * 5 * CAP, 256), 256, 0, stream>>>((unsigned*)cand_s, 4UL * 5 * CAP);

  const int   HWs[5] = {80, 40, 20, 10, 5};
  const float STR[5] = {8.f, 16.f, 32.f, 64.f, 128.f};

  for (int lvl = 0; lvl < 5; ++lvl) {
    int H = HWs[lvl], W = H;
    float stride = STR[lvl];
    int tilesX = (W + 15) / 16;
    size_t nact = 4UL * 256 * H * W;

    k_f32_to_f16<<<gridFor(nact, 256), 256, 0, stream>>>(feats[lvl], act0, nact);

    auto conv = [&](const _Float16* src, const _Float16* wt, const float* bias,
                    _Float16* dst, int Cout, int CoutPad, int relu) {
      int ntiles = CoutPad / 16;
      dim3 g((unsigned)(4 * H * tilesX), (unsigned)((ntiles + WPB - 1) / WPB));
      conv3x3_wmma<<<g, 32 * WPB, SMEM_BYTES, stream>>>(src, wt, bias, dst,
                                                        Cout, CoutPad, H, W, tilesX, relu);
    };

    // cls head
    conv(act0, w_cls_h + 0 * WLAYER, cls_bh + 0,   act1, 256, 256, 1);
    conv(act1, w_cls_h + 1 * WLAYER, cls_bh + 256, act2, 256, 256, 1);
    conv(act2, w_cls_h + 2 * WLAYER, cls_bh + 512, act1, 256, 256, 1);
    conv(act1, w_cls_h + 3 * WLAYER, cls_bh + 768, act2, 256, 256, 1);
    conv(act2, w_cls_o, cls_bo, cls_raw, 720, 720, 0);
    // box head
    conv(act0, w_box_h + 0 * WLAYER, box_bh + 0,   act1, 256, 256, 1);
    conv(act1, w_box_h + 1 * WLAYER, box_bh + 256, act2, 256, 256, 1);
    conv(act2, w_box_h + 2 * WLAYER, box_bh + 512, act1, 256, 256, 1);
    conv(act1, w_box_h + 3 * WLAYER, box_bh + 768, act2, 256, 256, 1);
    conv(act2, w_box_o, box_bo, box_raw, 36, 48, 0);

    // per-level top-k selection (histogram select) + decode
    int N1 = 720 * H * W;
    k_zero32<<<gridFor(4UL * NBIN + 4, 256), 256, 0, stream>>>(hist, 4UL * NBIN + 4); // hist + cnt
    k_hist<<<gridFor((size_t)4 * N1, 256), 256, 0, stream>>>(cls_raw, N1, hist);
    k_cutoff<<<1, 32, 0, stream>>>(hist, cutoff);
    k_collect<<<gridFor((size_t)4 * N1, 256), 256, 0, stream>>>(cls_raw, box_raw, cutoff, cnt,
                                                                cand_s, cand_b, cand_c,
                                                                H, W, lvl, stride);
  }

  k_nms<<<4, 256, 0, stream>>>(cand_s, cand_b, cand_c, (float*)d_out);
}
